// GlobalPointer_51496657879540
// MI455X (gfx1250) — matrix operand order repaired and checked by reading, not compiled, not run on hardware
//
#include <hip/hip_runtime.h>

// MI455X / gfx1250, wave32. Matrix ops via v_wmma_f32_16x16x32_f16.
typedef __attribute__((ext_vector_type(16))) _Float16 v16h;
typedef __attribute__((ext_vector_type(8)))  _Float16 v8h;
typedef __attribute__((ext_vector_type(8)))  float    v8f;

#define BATCH   8
#define SEQ     512
#define HEADS   24
#define DH      64
#define BDIM    768
#define NOUT    3072            // HEADS * DH * 2
#define NEG_INF 1000000000000.0f

static __device__ __forceinline__ v8f wmma_f16(v16h a, v16h b, v8f c) {
  // 8 args: (neg_a, A, neg_b, B, c_mod, C, reuse_a, reuse_b)
  return __builtin_amdgcn_wmma_f32_16x16x32_f16(false, a, false, b, (short)0, c, false, false);
}

// A-matrix 16x32 f16 fragment: lane lo holds K=k0+0..15; lane hi holds
// K=k0+16..31 (same contiguous order as B).
static __device__ __forceinline__ v16h load_a_frag(const _Float16* __restrict__ row,
                                                   int k0, int hi) {
  v8h lo  = *(const v8h*)(row + k0 + 16 * hi);
  v8h hi8 = *(const v8h*)(row + k0 + 16 * hi + 8);
  v16h a;
#pragma unroll
  for (int i = 0; i < 8; ++i) { a[i] = lo[i]; a[i + 8] = hi8[i]; }
  return a;
}

// B-matrix 32x16 f16 fragment: lane lo holds K=k0+0..15 for N=lane, lane hi
// K=k0+16..31 for N=lane-16.  `rowN` = contiguous K-run for column N
// (source stored N-major with K contiguous == transposed W / K^T).
static __device__ __forceinline__ v16h load_b_frag(const _Float16* __restrict__ rowN,
                                                   int k0, int hi) {
  return *(const v16h*)(rowN + k0 + 16 * hi);
}

// ---------------------------------------------------------------------------
// Kernel 0: stage f16 copies.  in_h[m][k] = inputs; wt_h[n][k] = W[k][n].
// ---------------------------------------------------------------------------
__global__ void __launch_bounds__(256) convert_kernel(
    const float* __restrict__ in, const float* __restrict__ W,
    _Float16* __restrict__ in_h, _Float16* __restrict__ wt_h) {
  const int idx = blockIdx.x * blockDim.x + threadIdx.x;
  const int TOTAL_IN = BATCH * SEQ * BDIM;  // 3,145,728
  const int TOTAL_W  = BDIM * NOUT;         // 2,359,296
  if (idx < TOTAL_IN) in_h[idx] = (_Float16)in[idx];
  if (idx < TOTAL_W) {
    const int k = idx / NOUT;
    const int n = idx % NOUT;               // coalesced read of W row k
    wt_h[(size_t)n * BDIM + k] = (_Float16)W[idx];
  }
}

// ---------------------------------------------------------------------------
// Kernel 1: x = inputs @ W + b, split q/k, RoPE on q, store f16 (b,h,s,d).
// One wave -> 32(M) x 64(N) tile: 2 A-frags x 4 B-frags = 8 WMMAs per K-step.
// ---------------------------------------------------------------------------
__global__ void __launch_bounds__(128) proj_rope_kernel(
    const _Float16* __restrict__ in_h, const _Float16* __restrict__ wt_h,
    const float* __restrict__ bias,
    _Float16* __restrict__ q_h, _Float16* __restrict__ k_h) {
  const int lane  = threadIdx.x & 31;
  const int hi    = lane >> 4;
  const int l15   = lane & 15;
  const int gwave = blockIdx.x * 4 + (threadIdx.x >> 5);  // 6144 waves
  const int mtile = gwave & 127;                          // 128 tiles of 32 rows
  const int ngrp  = gwave >> 7;                           // 48 groups of 64 cols
  const int m0 = mtile * 32;
  const int n0 = ngrp * 64;

  v8f acc0[4] = {v8f{}, v8f{}, v8f{}, v8f{}};   // rows m0   .. m0+15
  v8f acc1[4] = {v8f{}, v8f{}, v8f{}, v8f{}};   // rows m0+16.. m0+31
  const _Float16* arow0 = in_h + (size_t)(m0 + l15) * BDIM;
  const _Float16* arow1 = in_h + (size_t)(m0 + 16 + l15) * BDIM;

  for (int k0 = 0; k0 < BDIM; k0 += 32) {
    v16h a0 = load_a_frag(arow0, k0, hi);
    v16h a1 = load_a_frag(arow1, k0, hi);
#pragma unroll
    for (int j = 0; j < 4; ++j) {
      v16h b = load_b_frag(wt_h + (size_t)(n0 + j * 16 + l15) * BDIM, k0, hi);
      acc0[j] = wmma_f16(a0, b, acc0[j]);
      acc1[j] = wmma_f16(a1, b, acc1[j]);
    }
  }

  // Epilogue: bias add + RoPE on q half, store f16 (b,h,s,d).
#pragma unroll
  for (int j = 0; j < 4; ++j) {
    const int n      = n0 + j * 16 + l15;
    const int within = n & 127;                 // 16-col tiles never straddle 64/128 edges
    const int head   = n >> 7;
    const bool isQ   = within < DH;             // wave-uniform
    const int d      = isQ ? within : within - DH;
    const float bv   = bias[n];
    // inv_i = 10000^(-2i/64), i = d/2 ;  ln(10000)/32 = 0.28782313662425575
    const float inv  = __expf(-0.28782313662425575f * (float)(d >> 1));
    const float sgn  = (d & 1) ? 1.0f : -1.0f;
#pragma unroll
    for (int half = 0; half < 2; ++half) {
      const int mb = m0 + 16 * half;
#pragma unroll
      for (int r = 0; r < 8; ++r) {
        const int m  = mb + r + (hi << 3);      // C/D layout: hi lanes hold M=r+8
        const int bI = m >> 9;                  // m / SEQ
        const int s  = m & (SEQ - 1);
        float val = (half ? acc1[j][r] : acc0[j][r]) + bv;
        if (isQ) {
          // q'[2i]   = q[2i]  *cos_i - q[2i+1]*sin_i
          // q'[2i+1] = q[2i+1]*cos_i + q[2i]  *sin_i   (partner = adjacent lane)
          const float partner = __shfl_xor(val, 1, 32);
          float sn, c;
          __sincosf((float)s * inv, &sn, &c);
          val = val * c + sgn * partner * sn;
          q_h[((size_t)((bI * HEADS + head) * SEQ + s)) * DH + d] = (_Float16)val;
        } else {
          k_h[((size_t)((bI * HEADS + head) * SEQ + s)) * DH + d] = (_Float16)val;
        }
      }
    }
  }
}

// ---------------------------------------------------------------------------
// Kernel 2: logits[b,h,m,n] = sum_d q[m,d]*k[n,d], row/col mask, strict lower
// triangle -NEG_INF, scale 1/8.  One wave -> 32x64 tile, K=64 (2 steps).
// ---------------------------------------------------------------------------
__global__ void __launch_bounds__(128) attn_logits_kernel(
    const _Float16* __restrict__ q_h, const _Float16* __restrict__ k_h,
    const float* __restrict__ mask, float* __restrict__ out) {
  const int lane  = threadIdx.x & 31;
  const int hi    = lane >> 4;
  const int l15   = lane & 15;
  const int gwave = blockIdx.x * 4 + (threadIdx.x >> 5);  // 24576 waves
  const int bh    = gwave >> 7;                           // 192 (b*HEADS+h)
  const int t     = gwave & 127;
  const int mtile = t & 15;                               // 16 m-tiles of 32
  const int ngrp  = t >> 4;                               // 8 n-groups of 64
  const int bI    = bh / HEADS;
  const int m0 = mtile * 32;
  const int n0 = ngrp * 64;

  const _Float16* q  = q_h + (size_t)bh * SEQ * DH;
  const _Float16* kk = k_h + (size_t)bh * SEQ * DH;

  v8f acc0[4] = {v8f{}, v8f{}, v8f{}, v8f{}};
  v8f acc1[4] = {v8f{}, v8f{}, v8f{}, v8f{}};
  const _Float16* qrow0 = q + (size_t)(m0 + l15) * DH;
  const _Float16* qrow1 = q + (size_t)(m0 + 16 + l15) * DH;
#pragma unroll
  for (int k0 = 0; k0 < DH; k0 += 32) {
    v16h a0 = load_a_frag(qrow0, k0, hi);
    v16h a1 = load_a_frag(qrow1, k0, hi);
#pragma unroll
    for (int j = 0; j < 4; ++j) {
      // B[d][n] = k[n][d]: K^T columns are contiguous rows of k_h.
      v16h b = load_b_frag(kk + (size_t)(n0 + j * 16 + l15) * DH, k0, hi);
      acc0[j] = wmma_f16(a0, b, acc0[j]);
      acc1[j] = wmma_f16(a1, b, acc1[j]);
    }
  }

  const float* mrow = mask + bI * SEQ;
  float lncol[4];
#pragma unroll
  for (int j = 0; j < 4; ++j) lncol[j] = mrow[n0 + j * 16 + l15];

  float* obase = out + (size_t)bh * SEQ * SEQ;
#pragma unroll
  for (int half = 0; half < 2; ++half) {
    const int mb = m0 + 16 * half;
#pragma unroll
    for (int r = 0; r < 8; ++r) {
      const int m    = mb + r + (hi << 3);
      const float lm = mrow[m];
#pragma unroll
      for (int j = 0; j < 4; ++j) {
        const int n = n0 + j * 16 + l15;
        float v = half ? acc1[j][r] : acc0[j][r];
        v = v * lm       - NEG_INF * (1.0f - lm);        // row mask
        v = v * lncol[j] - NEG_INF * (1.0f - lncol[j]);  // col mask
        if (n < m) v -= NEG_INF;                         // tril(ones,-1)*NEG_INF
        v *= 0.125f;                                     // 1/sqrt(64)
        obase[(size_t)m * SEQ + n] = v;
      }
    }
  }
}

// ---------------------------------------------------------------------------
extern "C" void kernel_launch(void* const* d_in, const int* in_sizes, int n_in,
                              void* d_out, int out_size, void* d_ws, size_t ws_size,
                              hipStream_t stream) {
  (void)in_sizes; (void)n_in; (void)out_size; (void)ws_size;
  const float* inputs = (const float*)d_in[0];  // (8, 512, 768)
  const float* mask   = (const float*)d_in[1];  // (8, 512)
  const float* W      = (const float*)d_in[2];  // (768, 3072)
  const float* bias   = (const float*)d_in[3];  // (3072,)
  float* out = (float*)d_out;                   // (8, 24, 512, 512)

  char* ws = (char*)d_ws;
  size_t off = 0;
  _Float16* in_h = (_Float16*)(ws + off); off += (size_t)BATCH * SEQ * BDIM * 2;      // 6.29 MB
  _Float16* wt_h = (_Float16*)(ws + off); off += (size_t)BDIM * NOUT * 2;             // 4.72 MB
  _Float16* q_h  = (_Float16*)(ws + off); off += (size_t)BATCH * HEADS * SEQ * DH * 2;// 12.6 MB
  _Float16* k_h  = (_Float16*)(ws + off);                                             // 12.6 MB

  convert_kernel    <<<12288, 256, 0, stream>>>(inputs, W, in_h, wt_h);
  proj_rope_kernel  <<<1536, 128, 0, stream>>>(in_h, wt_h, bias, q_h, k_h);
  attn_logits_kernel<<<6144, 128, 0, stream>>>(q_h, k_h, mask, out);
}
